// LSTMRegressor_76751065579884
// MI455X (gfx1250) — matrix operand order repaired and checked by reading, not compiled, not run on hardware
//
#include <hip/hip_runtime.h>
#include <hip/hip_bf16.h>

// ---------------------------------------------------------------------------
// Types for CDNA5 WMMA operands
// ---------------------------------------------------------------------------
typedef __attribute__((ext_vector_type(16))) _Float16 v16h;
typedef __attribute__((ext_vector_type(16))) __bf16   v16bf;
typedef __attribute__((ext_vector_type(8)))  float    v8f;

union V16H { v16h  v; unsigned int u[8]; _Float16 h[16]; };
union V16B { v16bf v; unsigned int u[8]; unsigned short s[16]; };
union V8F  { v8f   v; float f[8]; };

#define LOG2E 1.44269504088896340736f

__device__ __forceinline__ unsigned short f32_to_bf16(float f) {
    unsigned int u = __builtin_bit_cast(unsigned int, f);
    u += 0x7FFFu + ((u >> 16) & 1u);          // round-to-nearest-even
    return (unsigned short)(u >> 16);
}

// Branch-free gate activation: so * rcp(1 + exp2(ck*v)) + off,  ck = -kx*log2e
//   sigmoid: kx=1, so=1, off=0      tanh: kx=2, so=2, off=-1
// Saturates cleanly: exp2 -> 0 => so+off; exp2 -> inf => rcp -> 0 => off.
// Lowers to v_mul + v_exp_f32 + v_add + v_rcp_f32 + v_fma (single exp scale).
__device__ __forceinline__ float gate_act(float v, float ck, float so, float off) {
    float e = __builtin_amdgcn_exp2f(ck * v);        // v_exp_f32
    float r = __builtin_amdgcn_rcpf(1.0f + e);       // v_rcp_f32
    return __builtin_fmaf(so, r, off);
}
__device__ __forceinline__ float fast_tanh(float x) {
    return gate_act(x, -2.0f * LOG2E, 2.0f, -1.0f);
}

// ---------------------------------------------------------------------------
// LSTM kernel: one block per 16 batch rows, 4 waves; wave w owns gate w
// (i=0, f=1, g=2, o=3; 32 gate columns each = 2 WMMA_f32_16x16x32_f16/step).
// A-operand = h tile (16x32 f16), B-operand = W_hh gate slice (32x16 f16),
// C-operand = x*x2h_w + x2h_b + h2h_b built per lane.
// ---------------------------------------------------------------------------
__global__ __launch_bounds__(128)
void lstm_kernel(const float* __restrict__ x,        // [B, T]  (x inner dim 1)
                 const float* __restrict__ x2h_w,    // [128]
                 const float* __restrict__ x2h_b,    // [128]
                 const float* __restrict__ h2h_w,    // [32,128]
                 const float* __restrict__ h2h_b,    // [128]
                 unsigned short* __restrict__ hs)    // bf16 [B, T*32]
{
    constexpr int T = 512, HID = 32, NG = 128;
    const int lane = threadIdx.x & 31;
    const int w    = threadIdx.x >> 5;   // gate id / wave id: 0..3
    const int hi   = lane >> 4;          // half-wave 0/1
    const int ln   = lane & 15;
    const int base = blockIdx.x * 16;    // batch row base

    __shared__ __align__(16) _Float16 hbuf[16][40];   // h tile, padded rows
    __shared__ float gbuf[4][16][33];                 // activated gates i,f,g,o
    __shared__ float xbuf[16][516];                   // x slice, padded rows

    for (int i = threadIdx.x; i < 16 * 40; i += 128)
        (&hbuf[0][0])[i] = (_Float16)0.0f;

    // stage x[base..base+16, 0..T) into LDS, coalesced
    for (int i = threadIdx.x; i < 16 * T; i += 128) {
        int r = i >> 9, t = i & 511;
        xbuf[r][t] = x[(size_t)(base + r) * T + t];
    }

    // B-operands: W_hh columns [32w + 16s, +16).  B layout (32x16, K=32):
    // lane l holds column N=l&15, half j holds K = (j&7) + 8*(l>>4) + 16*(j>>3).
    V16H Bop[2];
    float xw[2], bias[2];
    for (int s = 0; s < 2; ++s) {
        int n = w * 32 + s * 16 + ln;
        xw[s]   = x2h_w[n];
        bias[s] = x2h_b[n] + h2h_b[n];
        for (int j = 0; j < 16; ++j) {
            int k = (j & 7) + 8 * hi + 16 * (j >> 3);
            Bop[s].h[j] = (_Float16)h2h_w[k * NG + n];
        }
    }

    // wave-uniform activation constants (gate 2 = g uses tanh)
    const float ack  = (w == 2) ? -2.0f * LOG2E : -LOG2E;
    const float aso  = (w == 2) ? 2.0f : 1.0f;
    const float aoff = (w == 2) ? -1.0f : 0.0f;

    // c/h update role: this wave owns columns [8w, 8w+8); lane -> row ln,
    // 4 columns starting at 8w + 4*hi.
    const int mc   = ln;
    const int col0 = w * 8 + 4 * hi;
    float c[4] = {0.f, 0.f, 0.f, 0.f};

    __syncthreads();

    for (int t = 0; t < T; ++t) {
        // A-operand from hbuf (16x32 f16): lane row M=ln; VGPR r holds the
        // K pair K0 = 2*(r&3) + 16*(r>>2) + 8*hi  (ISA 7.12.2 table).
        V16H hA;
#pragma unroll
        for (int r = 0; r < 8; ++r) {
            int k0 = 2 * (r & 3) + 16 * (r >> 2) + 8 * hi;
            hA.u[r] = *(const unsigned int*)&hbuf[ln][k0];
        }

        float xv = xbuf[ln][t];

        V8F acc[2];
#pragma unroll
        for (int s = 0; s < 2; ++s) {
#pragma unroll
            for (int r = 0; r < 8; ++r) {
                int m = r + 8 * hi;                 // C row for this VGPR
                float xm = __shfl(xv, m, 32);       // x[base+m, t]
                acc[s].f[r] = __builtin_fmaf(xm, xw[s], bias[s]); // gx + biases
            }
            acc[s].v = __builtin_amdgcn_wmma_f32_16x16x32_f16(
                false, hA.v, false, Bop[s].v, (short)0, acc[s].v, false, false);
        }

        // activation of this wave's gate (branch-free), scatter to gbuf
#pragma unroll
        for (int s = 0; s < 2; ++s)
#pragma unroll
            for (int r = 0; r < 8; ++r) {
                float a = gate_act(acc[s].f[r], ack, aso, aoff);
                gbuf[w][r + 8 * hi][s * 16 + ln] = a;
            }
        __syncthreads();

        // cell/hidden update for 4 owned (row=mc, col) elements
#pragma unroll
        for (int e = 0; e < 4; ++e) {
            int col = col0 + e;
            float ig = gbuf[0][mc][col];
            float fg = gbuf[1][mc][col];
            float gg = gbuf[2][mc][col];
            float og = gbuf[3][mc][col];
            c[e] = __builtin_fmaf(c[e], fg, ig * gg);
            float hv = og * fast_tanh(c[e]);
            hbuf[mc][col] = (_Float16)hv;
            hs[(size_t)(base + mc) * (T * HID) + (size_t)t * HID + col] =
                f32_to_bf16(hv);
        }
        __syncthreads();
    }
}

// ---------------------------------------------------------------------------
// Tiled bf16 WMMA GEMM:  C = act(A[M,K](bf16) @ W[K,N](f32->bf16) + bias)
// Block tile 128x128, BK=32, 256 threads (8 waves, each 64x32 = 4x2 WMMAs).
// Software-pipelined: tile k+1 global loads issued before tile k compute.
// ---------------------------------------------------------------------------
template <bool RELU, bool OUT_BF16>
__global__ __launch_bounds__(256)
void gemm_bf16_kernel(const unsigned short* __restrict__ A,  // bf16 [M,K]
                      const float* __restrict__ W,           // f32 [K,N]
                      const float* __restrict__ bias,        // f32 [N]
                      void* __restrict__ Cout,               // [M,N]
                      int M, int N, int K)
{
    const int tid  = threadIdx.x;
    const int lane = tid & 31;
    const int wave = tid >> 5;
    const int hi   = lane >> 4;
    const int ln   = lane & 15;
    const int wm   = wave & 1;   // 0..1 -> 64-row slab
    const int wn   = wave >> 1;  // 0..3 -> 32-col slab

    __shared__ __align__(16) unsigned short As[128][40]; // [m][k], padded
    __shared__ __align__(16) unsigned short Bs[128][40]; // [n][k], padded

    const size_t mBase = (size_t)blockIdx.y * 128;
    const size_t nBase = (size_t)blockIdx.x * 128;

    const int arow  = tid >> 1, apart = tid & 1;        // A loader role
    const int brow  = tid >> 3, bcol  = (tid & 7) * 16; // W loader role

    V8F acc[4][2];
#pragma unroll
    for (int mt = 0; mt < 4; ++mt)
#pragma unroll
        for (int nt = 0; nt < 2; ++nt)
#pragma unroll
            for (int r = 0; r < 8; ++r) acc[mt][nt].f[r] = 0.0f;

    // ---- prologue: stage tile 0 into registers ----
    uint4 av = *(const uint4*)&A[(mBase + arow) * K + apart * 16];
    float wv[16];
    {
        const float* wp = &W[(size_t)brow * N + nBase + bcol];
#pragma unroll
        for (int q = 0; q < 4; ++q) {
            float4 t = *(const float4*)(wp + 4 * q);
            wv[4 * q + 0] = t.x; wv[4 * q + 1] = t.y;
            wv[4 * q + 2] = t.z; wv[4 * q + 3] = t.w;
        }
    }

    for (int kk = 0; kk < K; kk += 32) {
        __syncthreads();   // previous tile fully consumed
        *(uint4*)&As[arow][apart * 16] = av;           // 80B rows: 16B aligned
#pragma unroll
        for (int j = 0; j < 16; ++j)
            Bs[bcol + j][brow] = f32_to_bf16(wv[j]);   // transpose to [n][k]
        __syncthreads();

        // ---- issue next tile's global loads (latency hidden by WMMA) ----
        if (kk + 32 < K) {
            av = *(const uint4*)&A[(mBase + arow) * K + (kk + 32) + apart * 16];
            const float* wp = &W[(size_t)(kk + 32 + brow) * N + nBase + bcol];
#pragma unroll
            for (int q = 0; q < 4; ++q) {
                float4 t = *(const float4*)(wp + 4 * q);
                wv[4 * q + 0] = t.x; wv[4 * q + 1] = t.y;
                wv[4 * q + 2] = t.z; wv[4 * q + 3] = t.w;
            }
            if (kk + 64 < K)  // hint tile k+2 (global_prefetch_b8)
                __builtin_prefetch(&W[(size_t)(kk + 64 + brow) * N + nBase + bcol], 0, 1);
        }

        // ---- build operands & WMMA ----
        V16B aop[4];
#pragma unroll
        for (int mt = 0; mt < 4; ++mt) {
            int m = wm * 64 + mt * 16 + ln;
#pragma unroll
            for (int r = 0; r < 8; ++r) {
                int k0 = 2 * (r & 3) + 16 * (r >> 2) + 8 * hi; // A layout
                aop[mt].u[r] = *(const unsigned int*)&As[m][k0];
            }
        }
        V16B bop[2];
#pragma unroll
        for (int nt = 0; nt < 2; ++nt) {
            int n = wn * 32 + nt * 16 + ln;
#pragma unroll
            for (int r = 0; r < 8; ++r) {
                int k0 = 2 * (r & 3) + 16 * (r >> 2) + 8 * hi; // B layout
                bop[nt].u[r] = *(const unsigned int*)&Bs[n][k0];
            }
        }
#pragma unroll
        for (int mt = 0; mt < 4; ++mt)
#pragma unroll
            for (int nt = 0; nt < 2; ++nt)
                acc[mt][nt].v = __builtin_amdgcn_wmma_f32_16x16x32_bf16(
                    false, aop[mt].v, false, bop[nt].v,
                    (short)0, acc[mt][nt].v, false, false);
    }

    // ---- epilogue: bias (+ReLU), store ----
#pragma unroll
    for (int nt = 0; nt < 2; ++nt) {
        int coln = (int)nBase + wn * 32 + nt * 16 + ln;
        float bval = bias[coln];
#pragma unroll
        for (int mt = 0; mt < 4; ++mt) {
#pragma unroll
            for (int r = 0; r < 8; ++r) {
                int row = (int)mBase + wm * 64 + mt * 16 + r + 8 * hi;
                float vv = acc[mt][nt].f[r] + bval;
                if (RELU) vv = fmaxf(vv, 0.0f);
                size_t idx = (size_t)row * N + coln;
                if (OUT_BF16) ((unsigned short*)Cout)[idx] = f32_to_bf16(vv);
                else          ((float*)Cout)[idx] = vv;
            }
        }
    }
}

// ---------------------------------------------------------------------------
// fc3: out[b] = h2[b,:] . w + b   (256-wide dot, one wave per row)
// ---------------------------------------------------------------------------
__global__ __launch_bounds__(256)
void fc3_kernel(const float* __restrict__ h2, const float* __restrict__ w,
                const float* __restrict__ b, float* __restrict__ out)
{
    const int lane = threadIdx.x & 31;
    const int wave = threadIdx.x >> 5;
    const int row  = blockIdx.x * 8 + wave;
    float acc = 0.0f;
#pragma unroll
    for (int k = lane; k < 256; k += 32)
        acc = __builtin_fmaf(h2[(size_t)row * 256 + k], w[k], acc);
#pragma unroll
    for (int off = 16; off > 0; off >>= 1)
        acc += __shfl_xor(acc, off, 32);
    if (lane == 0) out[row] = acc + b[0];
}

// ---------------------------------------------------------------------------
// Launch
// ---------------------------------------------------------------------------
extern "C" void kernel_launch(void* const* d_in, const int* in_sizes, int n_in,
                              void* d_out, int out_size, void* d_ws, size_t ws_size,
                              hipStream_t stream)
{
    const float* x     = (const float*)d_in[0];
    const float* x2h_w = (const float*)d_in[1];
    const float* x2h_b = (const float*)d_in[2];
    const float* h2h_w = (const float*)d_in[3];
    const float* h2h_b = (const float*)d_in[4];
    const float* fc1_w = (const float*)d_in[5];
    const float* fc1_b = (const float*)d_in[6];
    const float* fc2_w = (const float*)d_in[7];
    const float* fc2_b = (const float*)d_in[8];
    const float* fc3_w = (const float*)d_in[9];
    const float* fc3_b = (const float*)d_in[10];
    float* out = (float*)d_out;

    constexpr int B = 1024, T = 512, HID = 32;
    char* ws = (char*)d_ws;
    unsigned short* hs = (unsigned short*)ws;                                  // bf16 [B, T*32]
    unsigned short* h1 = (unsigned short*)(ws + (size_t)B * T * HID * 2);      // bf16 [B,2048]
    float*          h2 = (float*)(ws + (size_t)B * T * HID * 2 + (size_t)B * 2048 * 2); // f32 [B,256]

    hipLaunchKernelGGL(lstm_kernel, dim3(B / 16), dim3(128), 0, stream,
                       x, x2h_w, x2h_b, h2h_w, h2h_b, hs);

    hipLaunchKernelGGL((gemm_bf16_kernel<true, true>),
                       dim3(2048 / 128, B / 128), dim3(256), 0, stream,
                       hs, fc1_w, fc1_b, (void*)h1, B, 2048, T * HID);

    hipLaunchKernelGGL((gemm_bf16_kernel<true, false>),
                       dim3(256 / 128, B / 128), dim3(256), 0, stream,
                       h1, fc2_w, fc2_b, (void*)h2, B, 256, 2048);

    hipLaunchKernelGGL(fc3_kernel, dim3(B / 8), dim3(256), 0, stream,
                       h2, fc3_w, fc3_b, out);
}
